// PRGAT2_34651796144593
// MI455X (gfx1250) — compile-verified, matrix operand order checked by reading, not run on hardware
//
#include <hip/hip_runtime.h>
#include <hip/hip_bf16.h>

#define NN 50000
#define NE 800000
#define NREL 8
#define NBASES 4
#define LEAKY 0.2f

typedef __attribute__((ext_vector_type(16))) __bf16 v16bf;
typedef __attribute__((ext_vector_type(8)))  float  v8f;

// CDNA5 16-bit A/B fragment K index for element i (0..15) given hi = lane>>4.
// i in 0..7  -> K = hi*8 + i
// i in 8..15 -> K = 16 + hi*8 + (i-8)
__device__ __forceinline__ int k_of(int i, int hi) {
  return (i < 8) ? (hi * 8 + i) : (16 + hi * 8 + (i - 8));
}

// Load one A fragment: lane's elements are row[kk+hi*8 .. +7] and row[kk+hi*8+16 .. +23],
// scaled by s and converted to bf16. Four b128 loads, packed cvt.
__device__ __forceinline__ v16bf load_a_frag(const float* __restrict__ q, float s) {
  float4 a0 = *(const float4*)(q);
  float4 a1 = *(const float4*)(q + 4);
  float4 a2 = *(const float4*)(q + 16);
  float4 a3 = *(const float4*)(q + 20);
  v16bf a;
  a[0]  = (__bf16)(a0.x * s); a[1]  = (__bf16)(a0.y * s);
  a[2]  = (__bf16)(a0.z * s); a[3]  = (__bf16)(a0.w * s);
  a[4]  = (__bf16)(a1.x * s); a[5]  = (__bf16)(a1.y * s);
  a[6]  = (__bf16)(a1.z * s); a[7]  = (__bf16)(a1.w * s);
  a[8]  = (__bf16)(a2.x * s); a[9]  = (__bf16)(a2.y * s);
  a[10] = (__bf16)(a2.z * s); a[11] = (__bf16)(a2.w * s);
  a[12] = (__bf16)(a3.x * s); a[13] = (__bf16)(a3.y * s);
  a[14] = (__bf16)(a3.z * s); a[15] = (__bf16)(a3.w * s);
  return a;
}

// ---------------- weight prep: pack into WMMA B-fragment layout ----------------
// Packed layout: [... chunk = kc*OCT+oc ...][lane 0..31][elem 0..15]
// where element (lane,i) = W[kc*32 + k_of(i,lane>>4)][oc*16 + (lane&15)]

__global__ void pack_combine_kernel(const float* __restrict__ comp,
                                    const float* __restrict__ basis,
                                    __bf16* __restrict__ out, int IN, int OUT) {
  int idx = blockIdx.x * blockDim.x + threadIdx.x;
  int PR = IN * OUT;
  if (idx >= NREL * PR) return;
  int r = idx / PR, f = idx % PR;
  int OCT = OUT / 16;
  int c = f >> 9;            // chunk index
  int t = f & 511;
  int lane = t >> 4, i = t & 15;
  int kc = c / OCT, oc = c % OCT;
  int k = kc * 32 + k_of(i, lane >> 4);
  int col = oc * 16 + (lane & 15);
  float acc = 0.f;
#pragma unroll
  for (int b = 0; b < NBASES; ++b)
    acc += comp[r * NBASES + b] * basis[(size_t)b * PR + (size_t)k * OUT + col];
  out[idx] = (__bf16)acc;
}

__global__ void pack_weight_kernel(const float* __restrict__ W, __bf16* __restrict__ out,
                                   int IN, int OUT) {
  int idx = blockIdx.x * blockDim.x + threadIdx.x;
  if (idx >= IN * OUT) return;
  int OCT = OUT / 16;
  int c = idx >> 9;
  int t = idx & 511;
  int lane = t >> 4, i = t & 15;
  int kc = c / OCT, oc = c % OCT;
  int k = kc * 32 + k_of(i, lane >> 4);
  int col = oc * 16 + (lane & 15);
  out[idx] = (__bf16)W[(size_t)k * OUT + col];
}

// ---------------- RGCN scatter ----------------

__global__ void count_kernel(const int* __restrict__ dst, const int* __restrict__ et,
                             float* __restrict__ cnt) {
  int e = blockIdx.x * blockDim.x + threadIdx.x;
  if (e >= NE) return;
  atomicAdd(&cnt[(size_t)dst[e] * NREL + et[e]], 1.0f);
}

__global__ void scatter_sums_kernel(const float* __restrict__ x, const int* __restrict__ src,
                                    const int* __restrict__ dst, const int* __restrict__ et,
                                    float* __restrict__ sums, int F) {
  int gid = blockIdx.x * blockDim.x + threadIdx.x;
  int e = gid >> 5, lane = gid & 31;
  if (e >= NE) return;
  size_t seg = (size_t)dst[e] * NREL + et[e];
  const float* xr = x + (size_t)src[e] * F;
  float* sr = sums + seg * F;
  for (int f = lane; f < F; f += 32) atomicAdd(&sr[f], xr[f]);
}

// ---------------- WMMA dense kernels ----------------

// out[16x16 tile] = relu( sum_r mean(n,r,:) @ Wrel[r] + x @ Wroot + bias )
template <int IN>
__global__ __launch_bounds__(32) void rgcn_dense_kernel(
    const float* __restrict__ sums, const float* __restrict__ cnt,
    const float* __restrict__ xin, const __bf16* __restrict__ Wrel,   // packed [R][KCH][OCT][32][16]
    const __bf16* __restrict__ Wroot,                                 // packed [KCH][OCT][32][16]
    const float* __restrict__ bias, float* __restrict__ out, int OUT) {
  constexpr int KCH = IN / 32;
  const int OCT = OUT / 16;
  const int lane = threadIdx.x;
  const int hi = lane >> 4, lo = lane & 15;
  const int nodeBase = blockIdx.x * 16;
  const int oc = blockIdx.y;
  const int ocol = oc * 16 + lo;
  const int rowNode = nodeBase + lo;
  v8f acc = {};
#pragma unroll
  for (int r = 0; r < NREL; ++r) {
    // mean = sums / max(cnt,1); hardware rcp (1 ulp) is exact enough: result is
    // immediately rounded to bf16 (8-bit mantissa) for the WMMA A operand.
    const float rc = __builtin_amdgcn_rcpf(fmaxf(cnt[(size_t)rowNode * NREL + r], 1.0f));
    const float* srow = sums + ((size_t)rowNode * NREL + r) * IN + hi * 8;
    const __bf16* wr = Wrel + ((((size_t)r * KCH) * OCT + oc) * 32 + lane) * 16;
#pragma unroll
    for (int kc = 0; kc < KCH; ++kc) {
      v16bf a = load_a_frag(srow + kc * 32, rc);
      v16bf b = *(const v16bf*)(wr + (size_t)kc * OCT * 512);
      acc = __builtin_amdgcn_wmma_f32_16x16x32_bf16(false, a, false, b, (short)0, acc, false, false);
    }
  }
  {
    const float* xrow = xin + (size_t)rowNode * IN + hi * 8;
    const __bf16* wr = Wroot + ((size_t)oc * 32 + lane) * 16;
#pragma unroll
    for (int kc = 0; kc < KCH; ++kc) {
      v16bf a = load_a_frag(xrow + kc * 32, 1.0f);
      v16bf b = *(const v16bf*)(wr + (size_t)kc * OCT * 512);
      acc = __builtin_amdgcn_wmma_f32_16x16x32_bf16(false, a, false, b, (short)0, acc, false, false);
    }
  }
  const float bv = bias[ocol];
#pragma unroll
  for (int v = 0; v < 8; ++v) {
    int m = v + 8 * hi;  // C/D row
    out[(size_t)(nodeBase + m) * OUT + ocol] = fmaxf(acc[v] + bv, 0.0f);
  }
}

// out = X @ W  (no bias / activation), W packed in fragment layout
template <int IN>
__global__ __launch_bounds__(32) void wmma_gemm_kernel(
    const float* __restrict__ X, const __bf16* __restrict__ W,
    float* __restrict__ out, int OUT) {
  constexpr int KCH = IN / 32;
  const int OCT = OUT / 16;
  const int lane = threadIdx.x;
  const int hi = lane >> 4, lo = lane & 15;
  const int nodeBase = blockIdx.x * 16;
  const int oc = blockIdx.y;
  const int ocol = oc * 16 + lo;
  const int rowNode = nodeBase + lo;
  v8f acc = {};
  const float* xrow = X + (size_t)rowNode * IN + hi * 8;
  const __bf16* wr = W + ((size_t)oc * 32 + lane) * 16;
#pragma unroll
  for (int kc = 0; kc < KCH; ++kc) {
    v16bf a = load_a_frag(xrow + kc * 32, 1.0f);
    v16bf b = *(const v16bf*)(wr + (size_t)kc * OCT * 512);
    acc = __builtin_amdgcn_wmma_f32_16x16x32_bf16(false, a, false, b, (short)0, acc, false, false);
  }
#pragma unroll
  for (int v = 0; v < 8; ++v) {
    int m = v + 8 * hi;
    out[(size_t)(nodeBase + m) * OUT + ocol] = acc[v];
  }
}

// ---------------- GAT kernels ----------------

__global__ void attn_logit_kernel(const float* __restrict__ h, const float* __restrict__ asrc,
                                  const float* __restrict__ adst, float* __restrict__ als,
                                  float* __restrict__ ald, int C) {
  int idx = blockIdx.x * blockDim.x + threadIdx.x;
  if (idx >= NN * 4) return;
  int n = idx >> 2, hh = idx & 3;
  const float* hp = h + ((size_t)n * 4 + hh) * C;
  float s = 0.f, d = 0.f;
  for (int c = 0; c < C; ++c) { s += hp[c] * asrc[hh * C + c]; d += hp[c] * adst[hh * C + c]; }
  als[idx] = s; ald[idx] = d;
}

__device__ __forceinline__ void atomicMaxF(float* addr, float val) {
  unsigned int* ua = (unsigned int*)addr;
  unsigned int old = *ua;
  while (__uint_as_float(old) < val) {
    unsigned int assumed = old;
    old = atomicCAS(ua, assumed, __float_as_uint(val));
    if (old == assumed) break;
  }
}

__global__ void fill_kernel(float* __restrict__ p, float v, int n) {
  int i = blockIdx.x * blockDim.x + threadIdx.x;
  if (i < n) p[i] = v;
}

__global__ void init_bias_kernel(float* __restrict__ acc, const float* __restrict__ bias, int C) {
  int i = blockIdx.x * blockDim.x + threadIdx.x;
  if (i < NN * C) acc[i] = bias[i % C];
}

__device__ __forceinline__ float leaky_e(const float* als, const float* ald, int s, int d, int h) {
  float v = als[s * 4 + h] + ald[d * 4 + h];
  return v < 0.f ? LEAKY * v : v;
}

__global__ void edge_max_kernel(const int* __restrict__ src, const int* __restrict__ dst,
                                const float* __restrict__ als, const float* __restrict__ ald,
                                float* __restrict__ m) {
  int e = blockIdx.x * blockDim.x + threadIdx.x;
  if (e >= NE + NN) return;
  int s, d;
  if (e < NE) { s = src[e]; d = dst[e]; } else { s = d = e - NE; }
#pragma unroll
  for (int h = 0; h < 4; ++h) atomicMaxF(&m[d * 4 + h], leaky_e(als, ald, s, d, h));
}

__global__ void edge_expsum_kernel(const int* __restrict__ src, const int* __restrict__ dst,
                                   const float* __restrict__ als, const float* __restrict__ ald,
                                   const float* __restrict__ m, float* __restrict__ z) {
  int e = blockIdx.x * blockDim.x + threadIdx.x;
  if (e >= NE + NN) return;
  int s, d;
  if (e < NE) { s = src[e]; d = dst[e]; } else { s = d = e - NE; }
#pragma unroll
  for (int h = 0; h < 4; ++h)
    atomicAdd(&z[d * 4 + h], __expf(leaky_e(als, ald, s, d, h) - m[d * 4 + h]));
}

__global__ void edge_scatter_kernel(const float* __restrict__ hfeat, const int* __restrict__ src,
                                    const int* __restrict__ dst, const float* __restrict__ als,
                                    const float* __restrict__ ald, const float* __restrict__ m,
                                    const float* __restrict__ z, float* __restrict__ acc,
                                    int HC, int C) {
  int gid = blockIdx.x * blockDim.x + threadIdx.x;
  int e = gid >> 5, lane = gid & 31;
  if (e >= NE + NN) return;
  int s, d;
  if (e < NE) { s = src[e]; d = dst[e]; } else { s = d = e - NE; }
  float alpha[4];
#pragma unroll
  for (int hh = 0; hh < 4; ++hh)
    alpha[hh] = __expf(leaky_e(als, ald, s, d, hh) - m[d * 4 + hh]) *
                __builtin_amdgcn_rcpf(z[d * 4 + hh]);
  for (int ch = lane; ch < HC; ch += 32) {
    int hh = ch / C;
    atomicAdd(&acc[(size_t)d * HC + ch], hfeat[(size_t)s * HC + ch] * alpha[hh]);
  }
}

__global__ void final_kernel(const float* __restrict__ acc, const float* __restrict__ bias,
                             float* __restrict__ out) {
  int idx = blockIdx.x * blockDim.x + threadIdx.x;
  if (idx >= NN * 16) return;
  int n = idx >> 4, c = idx & 15;
  float v = 0.f;
#pragma unroll
  for (int h = 0; h < 4; ++h) v += acc[(size_t)n * 64 + h * 16 + c];
  v = v * 0.25f + bias[c];
  v = fmaxf(v, 0.f);
  out[idx] = tanhf(v);
}

// ---------------- host side ----------------

static inline int cdiv(long a, long b) { return (int)((a + b - 1) / b); }

extern "C" void kernel_launch(void* const* d_in, const int* in_sizes, int n_in,
                              void* d_out, int out_size, void* d_ws, size_t ws_size,
                              hipStream_t stream) {
  const float* x      = (const float*)d_in[0];
  const int*   eidx   = (const int*)d_in[1];
  const int*   esrc   = eidx;
  const int*   edst   = eidx + NE;
  const int*   etype  = (const int*)d_in[2];
  const float* basis1 = (const float*)d_in[3];
  const float* comp1  = (const float*)d_in[4];
  const float* root1  = (const float*)d_in[5];
  const float* brg1   = (const float*)d_in[6];
  const float* wg1    = (const float*)d_in[7];
  const float* asrc1  = (const float*)d_in[8];
  const float* adst1  = (const float*)d_in[9];
  const float* bg1    = (const float*)d_in[10];
  const float* basis2 = (const float*)d_in[11];
  const float* comp2  = (const float*)d_in[12];
  const float* root2  = (const float*)d_in[13];
  const float* brg2   = (const float*)d_in[14];
  const float* wg2    = (const float*)d_in[15];
  const float* asrc2  = (const float*)d_in[16];
  const float* adst2  = (const float*)d_in[17];
  const float* bg2    = (const float*)d_in[18];
  float* out = (float*)d_out;

  // workspace carve
  char* p = (char*)d_ws;
  auto carve = [&](size_t bytes) -> char* {
    char* r = p;
    p += (bytes + 255) & ~(size_t)255;
    return r;
  };
  float*  sums   = (float*)carve((size_t)NN * NREL * 128 * 4);  // shared by both RGCN layers
  float*  cnt    = (float*)carve((size_t)NN * NREL * 4);
  __bf16* W1bf   = (__bf16*)carve((size_t)NREL * 64 * 64 * 2);
  __bf16* r1bf   = (__bf16*)carve((size_t)64 * 64 * 2);
  __bf16* wg1bf  = (__bf16*)carve((size_t)64 * 128 * 2);
  __bf16* W2bf   = (__bf16*)carve((size_t)NREL * 128 * 32 * 2);
  __bf16* r2bf   = (__bf16*)carve((size_t)128 * 32 * 2);
  __bf16* wg2bf  = (__bf16*)carve((size_t)32 * 64 * 2);
  float*  h1     = (float*)carve((size_t)NN * 64 * 4);
  float*  g1     = (float*)carve((size_t)NN * 128 * 4);
  float*  als    = (float*)carve((size_t)NN * 4 * 4);
  float*  ald    = (float*)carve((size_t)NN * 4 * 4);
  float*  mbuf   = (float*)carve((size_t)NN * 4 * 4);
  float*  zbuf   = (float*)carve((size_t)NN * 4 * 4);
  float*  h2     = (float*)carve((size_t)NN * 128 * 4);
  float*  h3     = (float*)carve((size_t)NN * 32 * 4);
  float*  g2     = (float*)carve((size_t)NN * 64 * 4);
  float*  acc2   = (float*)carve((size_t)NN * 64 * 4);

  const int BT = 256;
  const int E2 = NE + NN;

  // ---- weight prep (packed into WMMA fragment layout) ----
  pack_combine_kernel<<<cdiv(NREL * 64 * 64, BT), BT, 0, stream>>>(comp1, basis1, W1bf, 64, 64);
  pack_combine_kernel<<<cdiv(NREL * 128 * 32, BT), BT, 0, stream>>>(comp2, basis2, W2bf, 128, 32);
  pack_weight_kernel<<<cdiv(64 * 64, BT), BT, 0, stream>>>(root1, r1bf, 64, 64);
  pack_weight_kernel<<<cdiv(128 * 32, BT), BT, 0, stream>>>(root2, r2bf, 128, 32);
  pack_weight_kernel<<<cdiv(64 * 128, BT), BT, 0, stream>>>(wg1, wg1bf, 64, 128);
  pack_weight_kernel<<<cdiv(32 * 64, BT), BT, 0, stream>>>(wg2, wg2bf, 32, 64);

  // ---- RGCN layer 1 ----
  hipMemsetAsync(cnt, 0, (size_t)NN * NREL * 4, stream);
  hipMemsetAsync(sums, 0, (size_t)NN * NREL * 64 * 4, stream);
  count_kernel<<<cdiv(NE, BT), BT, 0, stream>>>(edst, etype, cnt);
  scatter_sums_kernel<<<cdiv((long)NE * 32, BT), BT, 0, stream>>>(x, esrc, edst, etype, sums, 64);
  {
    dim3 g(NN / 16, 64 / 16);
    rgcn_dense_kernel<64><<<g, 32, 0, stream>>>(sums, cnt, x, W1bf, r1bf, brg1, h1, 64);
  }

  // ---- GAT layer 1 (64 -> 4x32 concat) ----
  {
    dim3 g(NN / 16, 128 / 16);
    wmma_gemm_kernel<64><<<g, 32, 0, stream>>>(h1, wg1bf, g1, 128);
  }
  attn_logit_kernel<<<cdiv(NN * 4, BT), BT, 0, stream>>>(g1, asrc1, adst1, als, ald, 32);
  fill_kernel<<<cdiv(NN * 4, BT), BT, 0, stream>>>(mbuf, -1e30f, NN * 4);
  hipMemsetAsync(zbuf, 0, (size_t)NN * 4 * 4, stream);
  edge_max_kernel<<<cdiv(E2, BT), BT, 0, stream>>>(esrc, edst, als, ald, mbuf);
  edge_expsum_kernel<<<cdiv(E2, BT), BT, 0, stream>>>(esrc, edst, als, ald, mbuf, zbuf);
  init_bias_kernel<<<cdiv(NN * 128, BT), BT, 0, stream>>>(h2, bg1, 128);
  edge_scatter_kernel<<<cdiv((long)E2 * 32, BT), BT, 0, stream>>>(g1, esrc, edst, als, ald, mbuf,
                                                                  zbuf, h2, 128, 32);

  // ---- RGCN layer 2 (128 -> 32) ----
  hipMemsetAsync(sums, 0, (size_t)NN * NREL * 128 * 4, stream);
  scatter_sums_kernel<<<cdiv((long)NE * 32, BT), BT, 0, stream>>>(h2, esrc, edst, etype, sums, 128);
  {
    dim3 g(NN / 16, 32 / 16);
    rgcn_dense_kernel<128><<<g, 32, 0, stream>>>(sums, cnt, h2, W2bf, r2bf, brg2, h3, 32);
  }

  // ---- GAT layer 2 (32 -> 4x16, mean heads) ----
  {
    dim3 g(NN / 16, 64 / 16);
    wmma_gemm_kernel<32><<<g, 32, 0, stream>>>(h3, wg2bf, g2, 64);
  }
  attn_logit_kernel<<<cdiv(NN * 4, BT), BT, 0, stream>>>(g2, asrc2, adst2, als, ald, 16);
  fill_kernel<<<cdiv(NN * 4, BT), BT, 0, stream>>>(mbuf, -1e30f, NN * 4);
  hipMemsetAsync(zbuf, 0, (size_t)NN * 4 * 4, stream);
  edge_max_kernel<<<cdiv(E2, BT), BT, 0, stream>>>(esrc, edst, als, ald, mbuf);
  edge_expsum_kernel<<<cdiv(E2, BT), BT, 0, stream>>>(esrc, edst, als, ald, mbuf, zbuf);
  hipMemsetAsync(acc2, 0, (size_t)NN * 64 * 4, stream);
  edge_scatter_kernel<<<cdiv((long)E2 * 32, BT), BT, 0, stream>>>(g2, esrc, edst, als, ald, mbuf,
                                                                  zbuf, acc2, 64, 16);
  final_kernel<<<cdiv(NN * 16, BT), BT, 0, stream>>>(acc2, bg2, out);
  (void)in_sizes; (void)n_in; (void)out_size; (void)ws_size;
}
